// CoMPT_70987219468553
// MI455X (gfx1250) — matrix operand order repaired and checked by reading, not compiled, not run on hardware
//
#include <hip/hip_runtime.h>

// ---------------------------------------------------------------------------
// CoMPT graph-attention forward for MI455X (gfx1250), wave32 + bf16 WMMA.
// ---------------------------------------------------------------------------

constexpr int NN = 50000;   // nodes
constexpr int EE = 800000;  // edges
constexpr int DD = 128;     // hidden features
constexpr int HH = 8;       // heads (DH = 16 == one WMMA N-tile)

typedef __attribute__((ext_vector_type(16))) __bf16 bf16x16;
typedef __attribute__((ext_vector_type(2)))  __bf16 bf16x2;
typedef __attribute__((ext_vector_type(8)))  float  f32x8;

union FragB16 {
  unsigned int u[8];
  bf16x16 v;
};

// Pack two f32 into one dword of bf16 (RNE) -- native v_cvt_pk_bf16_f32 when
// available, else scalar fptrunc (also native bf16 on gfx1250).
__device__ __forceinline__ unsigned int pk_bf16(float a, float b) {
#if __has_builtin(__builtin_amdgcn_cvt_pk_bf16_f32)
  union { bf16x2 v; unsigned int u; } cv;
  cv.v = __builtin_amdgcn_cvt_pk_bf16_f32(a, b);
  return cv.u;
#else
  union { __bf16 h[2]; unsigned int u; } cv;
  cv.h[0] = (__bf16)a;
  cv.h[1] = (__bf16)b;
  return cv.u;
#endif
}

__device__ __forceinline__ unsigned short bf16_1(float a) {
  return (unsigned short)(pk_bf16(a, 0.0f) & 0xffffu);
}

// Load 8 consecutive fp32, convert to 4 packed-bf16 dwords.
__device__ __forceinline__ void lda8(const float* p, unsigned int* o) {
  float4 x = *(const float4*)(p);
  float4 y = *(const float4*)(p + 4);
  o[0] = pk_bf16(x.x, x.y);
  o[1] = pk_bf16(x.z, x.w);
  o[2] = pk_bf16(y.x, y.y);
  o[3] = pk_bf16(y.z, y.w);
}

// B fragment for D = A * W^T: B(K,N) = W[N][K].
// Lane n (0..15): column N = n, K = kk*32 + {0..7,16..23} ; lanes 16..31: K = +8.
__device__ __forceinline__ bf16x16 ldB(const unsigned short* __restrict__ Wb,
                                       int nt, int kk, int row, int half) {
  const uint4* p = (const uint4*)(Wb + (nt * 16 + row) * DD + kk * 32 + half * 8);
  uint4 a = p[0];
  uint4 b = p[2];
  FragB16 f;
  f.u[0] = a.x; f.u[1] = a.y; f.u[2] = a.z; f.u[3] = a.w;
  f.u[4] = b.x; f.u[5] = b.y; f.u[6] = b.z; f.u[7] = b.w;
  return f.v;
}

__device__ __forceinline__ f32x8 wmma_bf16(bf16x16 a, bf16x16 b, f32x8 c) {
  return __builtin_amdgcn_wmma_f32_16x16x32_bf16(false, a, false, b, (short)0, c,
                                                 false, false);
}

// Float atomic max via signed/unsigned integer ordering trick (sign-safe).
__device__ __forceinline__ void atomicMaxF(float* addr, float val) {
  if (val >= 0.0f)
    atomicMax((int*)addr, __float_as_int(val));
  else
    atomicMin((unsigned int*)addr, __float_as_uint(val));
}

__device__ __forceinline__ float mishf(float x) {
  float sp = (x > 20.0f) ? x : log1pf(expf(x));
  return x * tanhf(sp);
}

// ---------------------------------------------------------------------------
// Kernel 1: weights -> bf16; init max/sum/score-accumulator arrays.
// ---------------------------------------------------------------------------
__global__ void __launch_bounds__(256) compt_init_kernel(
    const float* __restrict__ Wq, const float* __restrict__ Wk,
    const float* __restrict__ Wv, const float* __restrict__ Wo,
    unsigned short* __restrict__ Wqb, unsigned short* __restrict__ Wkb,
    unsigned short* __restrict__ Wvb, unsigned short* __restrict__ Wob,
    float* __restrict__ m0, float* __restrict__ m1, float* __restrict__ m2,
    float* __restrict__ s0, float* __restrict__ s1, float* __restrict__ s2,
    float* __restrict__ S) {
  int i = blockIdx.x * blockDim.x + threadIdx.x;
  if (i < DD * DD) {
    Wqb[i] = bf16_1(Wq[i]);
    Wkb[i] = bf16_1(Wk[i]);
    Wvb[i] = bf16_1(Wv[i]);
    Wob[i] = bf16_1(Wo[i]);
  }
  if (i < NN * HH) {
    float ninf = -__builtin_inff();
    m0[i] = ninf; m1[i] = ninf; m2[i] = ninf;
    s0[i] = 0.0f; s1[i] = 0.0f; s2[i] = 0.0f;
    S[i] = 0.0f;
  }
}

// ---------------------------------------------------------------------------
// Kernel 2: q = h_node@Wq^T+bq, v = h_node@Wv^T+bv  (one wave = 16 node rows)
// ---------------------------------------------------------------------------
__global__ void __launch_bounds__(128) compt_node_qv_kernel(
    const float* __restrict__ h_node,
    const unsigned short* __restrict__ Wqb, const float* __restrict__ bq,
    const unsigned short* __restrict__ Wvb, const float* __restrict__ bv,
    float* __restrict__ q, float* __restrict__ v) {
  int wave = threadIdx.x >> 5;
  int tile = blockIdx.x * 4 + wave;
  if (tile >= NN / 16) return;
  int lane = threadIdx.x & 31;
  int row = lane & 15, half = lane >> 4;
  int r0 = tile * 16;

  FragB16 A[4];
  const float* hp = h_node + (size_t)(r0 + row) * DD;
#pragma unroll
  for (int kk = 0; kk < 4; ++kk) {
    int kb = kk * 32 + half * 8;
    lda8(hp + kb, &A[kk].u[0]);
    lda8(hp + kb + 16, &A[kk].u[4]);
  }
#pragma unroll
  for (int nt = 0; nt < 8; ++nt) {
    // Batch all B-fragment loads for this output tile, then run WMMAs.
    bf16x16 Bq[4], Bv[4];
#pragma unroll
    for (int kk = 0; kk < 4; ++kk) {
      Bq[kk] = ldB(Wqb, nt, kk, row, half);
      Bv[kk] = ldB(Wvb, nt, kk, row, half);
    }
    float b0 = bq[nt * 16 + row];
    float b1 = bv[nt * 16 + row];
    f32x8 aq = {b0, b0, b0, b0, b0, b0, b0, b0};
    f32x8 av = {b1, b1, b1, b1, b1, b1, b1, b1};
#pragma unroll
    for (int kk = 0; kk < 4; ++kk) {
      aq = wmma_bf16(A[kk].v, Bq[kk], aq);
      av = wmma_bf16(A[kk].v, Bv[kk], av);
    }
#pragma unroll
    for (int r = 0; r < 8; ++r) {
      int m = r + half * 8;  // C/D layout: VGPR r -> M=r (lanes<16) / r+8
      q[(size_t)(r0 + m) * DD + nt * 16 + row] = aq[r];
      v[(size_t)(r0 + m) * DD + nt * 16 + row] = av[r];
    }
  }
}

// ---------------------------------------------------------------------------
// Kernel 3: fused k = h_edge@Wk^T+bk (WMMA -> LDS) + per-head edge dots,
// writes 3 logit arrays [E,H] and atomic-max per (dst,head).
// ---------------------------------------------------------------------------
__global__ void __launch_bounds__(128) compt_edge_logits_kernel(
    const float* __restrict__ h_edge,
    const unsigned short* __restrict__ Wkb, const float* __restrict__ bk,
    const float* __restrict__ q,
    const int* __restrict__ src, const int* __restrict__ dst,
    float* __restrict__ l0, float* __restrict__ l1, float* __restrict__ l2,
    float* __restrict__ m0, float* __restrict__ m1, float* __restrict__ m2) {
  __shared__ float kbuf[4][16 * 132];  // 16 edges x 128 feat, padded stride
  int wave = threadIdx.x >> 5;
  int tile = blockIdx.x * 4 + wave;  // EE/16 divisible by 4: no guard needed
  int lane = threadIdx.x & 31;
  int row = lane & 15, half = lane >> 4;
  int e0 = tile * 16;
  float* kb_w = kbuf[wave];

  FragB16 A[4];
  const float* hp = h_edge + (size_t)(e0 + row) * DD;
#pragma unroll
  for (int kk = 0; kk < 4; ++kk) {
    int kb = kk * 32 + half * 8;
    lda8(hp + kb, &A[kk].u[0]);
    lda8(hp + kb + 16, &A[kk].u[4]);
  }
#pragma unroll
  for (int nt = 0; nt < 8; ++nt) {  // nt == head (DH = 16 = one N tile)
    bf16x16 Bk[4];
#pragma unroll
    for (int kk = 0; kk < 4; ++kk)
      Bk[kk] = ldB(Wkb, nt, kk, row, half);
    float b0 = bk[nt * 16 + row];
    f32x8 acc = {b0, b0, b0, b0, b0, b0, b0, b0};
#pragma unroll
    for (int kk = 0; kk < 4; ++kk)
      acc = wmma_bf16(A[kk].v, Bk[kk], acc);
#pragma unroll
    for (int r = 0; r < 8; ++r)
      kb_w[(r + half * 8) * 132 + nt * 16 + row] = acc[r];
  }
  __syncthreads();

  // 128 (edge,head) pairs per tile; each lane does 4.
#pragma unroll
  for (int pp = 0; pp < 4; ++pp) {
    int p = lane + pp * 32;
    int e = p >> 3, h = p & 7;
    int ge = e0 + e;
    int sn = src[ge], dn = dst[ge];
    const float4* qs = (const float4*)(q + (size_t)sn * DD + h * 16);
    const float4* qd = (const float4*)(q + (size_t)dn * DD + h * 16);
    const float4* kp = (const float4*)(kb_w + e * 132 + h * 16);
    float din = 0.0f, dou = 0.0f, dia = 0.0f;
#pragma unroll
    for (int t = 0; t < 4; ++t) {
      float4 a = qs[t], b = qd[t], c = kp[t];
      din += a.x * c.x + a.y * c.y + a.z * c.z + a.w * c.w;
      dou += b.x * c.x + b.y * c.y + b.z * c.z + b.w * c.w;
      dia += a.x * b.x + a.y * b.y + a.z * b.z + a.w * b.w;
    }
    din *= 0.25f; dou *= 0.25f; dia *= 0.25f;  // 1/sqrt(DH)
    size_t li = (size_t)ge * HH + h;
    l0[li] = din; l1[li] = dou; l2[li] = dia;
    int mi = dn * HH + h;
    atomicMaxF(m0 + mi, din);
    atomicMaxF(m1 + mi, dou);
    atomicMaxF(m2 + mi, dia);
  }
}

// ---------------------------------------------------------------------------
// Kernel 4: softmax denominators (segment sums of exp(l - m)).
// ---------------------------------------------------------------------------
__global__ void __launch_bounds__(256) compt_expsum_kernel(
    const int* __restrict__ dst,
    const float* __restrict__ l0, const float* __restrict__ l1, const float* __restrict__ l2,
    const float* __restrict__ m0, const float* __restrict__ m1, const float* __restrict__ m2,
    float* __restrict__ s0, float* __restrict__ s1, float* __restrict__ s2) {
  int i = blockIdx.x * blockDim.x + threadIdx.x;
  if (i >= EE * HH) return;
  int e = i >> 3, h = i & 7;
  int mi = dst[e] * HH + h;
  atomicAdd(s0 + mi, expf(l0[i] - m0[mi]));
  atomicAdd(s1 + mi, expf(l1[i] - m1[mi]));
  atomicAdd(s2 + mi, expf(l2[i] - m2[mi]));
}

// ---------------------------------------------------------------------------
// Kernel 5: score = (sm1+sm2+sm3) * distance^lam, accumulate S[dst,h].
// (v[dst] is segment-constant -> agg = S[n,h] * v[n,h,:] later.)
// ---------------------------------------------------------------------------
__global__ void __launch_bounds__(256) compt_score_kernel(
    const int* __restrict__ dst, const float* __restrict__ distance,
    const float* __restrict__ lam,
    const float* __restrict__ l0, const float* __restrict__ l1, const float* __restrict__ l2,
    const float* __restrict__ m0, const float* __restrict__ m1, const float* __restrict__ m2,
    const float* __restrict__ s0, const float* __restrict__ s1, const float* __restrict__ s2,
    float* __restrict__ S) {
  int i = blockIdx.x * blockDim.x + threadIdx.x;
  if (i >= EE * HH) return;
  int e = i >> 3, h = i & 7;
  int mi = dst[e] * HH + h;
  float p = expf(l0[i] - m0[mi]) / (s0[mi] + 1e-12f)
          + expf(l1[i] - m1[mi]) / (s1[mi] + 1e-12f)
          + expf(l2[i] - m2[mi]) / (s2[mi] + 1e-12f);
  float att = exp2f(lam[0] * log2f(distance[e]));  // d in [0.1,1] > 0
  atomicAdd(S + mi, p * att);
}

// ---------------------------------------------------------------------------
// Kernel 6: out = mish( (S .* v) @ Wo^T + bo )
// ---------------------------------------------------------------------------
__global__ void __launch_bounds__(128) compt_out_kernel(
    const float* __restrict__ v, const float* __restrict__ S,
    const unsigned short* __restrict__ Wob, const float* __restrict__ bo,
    float* __restrict__ out) {
  int wave = threadIdx.x >> 5;
  int tile = blockIdx.x * 4 + wave;
  if (tile >= NN / 16) return;
  int lane = threadIdx.x & 31;
  int row = lane & 15, half = lane >> 4;
  int r0 = tile * 16;
  int node = r0 + row;
  const float* vp = v + (size_t)node * DD;
  const float* Sp = S + node * HH;

  FragB16 A[4];
#pragma unroll
  for (int kk = 0; kk < 4; ++kk) {
    int kb = kk * 32 + half * 8;       // 8-float chunk stays inside one head
    float sA = Sp[kb >> 4];
    float sB = Sp[(kb + 16) >> 4];
    float4 x = *(const float4*)(vp + kb);
    float4 y = *(const float4*)(vp + kb + 4);
    A[kk].u[0] = pk_bf16(x.x * sA, x.y * sA);
    A[kk].u[1] = pk_bf16(x.z * sA, x.w * sA);
    A[kk].u[2] = pk_bf16(y.x * sA, y.y * sA);
    A[kk].u[3] = pk_bf16(y.z * sA, y.w * sA);
    float4 x2 = *(const float4*)(vp + kb + 16);
    float4 y2 = *(const float4*)(vp + kb + 20);
    A[kk].u[4] = pk_bf16(x2.x * sB, x2.y * sB);
    A[kk].u[5] = pk_bf16(x2.z * sB, x2.w * sB);
    A[kk].u[6] = pk_bf16(y2.x * sB, y2.y * sB);
    A[kk].u[7] = pk_bf16(y2.z * sB, y2.w * sB);
  }
#pragma unroll
  for (int nt = 0; nt < 8; ++nt) {
    bf16x16 Bo[4];
#pragma unroll
    for (int kk = 0; kk < 4; ++kk)
      Bo[kk] = ldB(Wob, nt, kk, row, half);
    float b0 = bo[nt * 16 + row];
    f32x8 acc = {b0, b0, b0, b0, b0, b0, b0, b0};
#pragma unroll
    for (int kk = 0; kk < 4; ++kk)
      acc = wmma_bf16(A[kk].v, Bo[kk], acc);
#pragma unroll
    for (int r = 0; r < 8; ++r)
      out[(size_t)(r0 + r + half * 8) * DD + nt * 16 + row] = mishf(acc[r]);
  }
}

// ---------------------------------------------------------------------------
extern "C" void kernel_launch(void* const* d_in, const int* in_sizes, int n_in,
                              void* d_out, int out_size, void* d_ws, size_t ws_size,
                              hipStream_t stream) {
  (void)in_sizes; (void)n_in; (void)out_size; (void)ws_size;
  const float* h_node   = (const float*)d_in[0];
  const float* h_edge   = (const float*)d_in[1];
  const float* distance = (const float*)d_in[2];
  const float* Wq = (const float*)d_in[3];
  const float* bq = (const float*)d_in[4];
  const float* Wk = (const float*)d_in[5];
  const float* bk = (const float*)d_in[6];
  const float* Wv = (const float*)d_in[7];
  const float* bv = (const float*)d_in[8];
  const float* Wo = (const float*)d_in[9];
  const float* bo = (const float*)d_in[10];
  const float* lam = (const float*)d_in[11];
  const int* src = (const int*)d_in[12];
  const int* dst = (const int*)d_in[13];
  float* out = (float*)d_out;

  // Workspace carve-up (256B-aligned): ~139 MB total.
  char* ws = (char*)d_ws;
  size_t off = 0;
  auto take = [&](size_t bytes) -> void* {
    void* p = (void*)(ws + off);
    off += (bytes + 255) & ~(size_t)255;
    return p;
  };
  float* q  = (float*)take((size_t)NN * DD * sizeof(float));
  float* vv = (float*)take((size_t)NN * DD * sizeof(float));
  unsigned short* Wqb = (unsigned short*)take((size_t)DD * DD * 2);
  unsigned short* Wkb = (unsigned short*)take((size_t)DD * DD * 2);
  unsigned short* Wvb = (unsigned short*)take((size_t)DD * DD * 2);
  unsigned short* Wob = (unsigned short*)take((size_t)DD * DD * 2);
  float* l0 = (float*)take((size_t)EE * HH * sizeof(float));
  float* l1 = (float*)take((size_t)EE * HH * sizeof(float));
  float* l2 = (float*)take((size_t)EE * HH * sizeof(float));
  float* m0 = (float*)take((size_t)NN * HH * sizeof(float));
  float* m1 = (float*)take((size_t)NN * HH * sizeof(float));
  float* m2 = (float*)take((size_t)NN * HH * sizeof(float));
  float* s0 = (float*)take((size_t)NN * HH * sizeof(float));
  float* s1 = (float*)take((size_t)NN * HH * sizeof(float));
  float* s2 = (float*)take((size_t)NN * HH * sizeof(float));
  float* S  = (float*)take((size_t)NN * HH * sizeof(float));

  compt_init_kernel<<<(NN * HH + 255) / 256, 256, 0, stream>>>(
      Wq, Wk, Wv, Wo, Wqb, Wkb, Wvb, Wob, m0, m1, m2, s0, s1, s2, S);

  compt_node_qv_kernel<<<(NN / 16 + 3) / 4, 128, 0, stream>>>(
      h_node, Wqb, bq, Wvb, bv, q, vv);

  compt_edge_logits_kernel<<<EE / 16 / 4, 128, 0, stream>>>(
      h_edge, Wkb, bk, q, src, dst, l0, l1, l2, m0, m1, m2);

  compt_expsum_kernel<<<(EE * HH + 255) / 256, 256, 0, stream>>>(
      dst, l0, l1, l2, m0, m1, m2, s0, s1, s2);

  compt_score_kernel<<<(EE * HH + 255) / 256, 256, 0, stream>>>(
      dst, distance, lam, l0, l1, l2, m0, m1, m2, s0, s1, s2, S);

  compt_out_kernel<<<(NN / 16 + 3) / 4, 128, 0, stream>>>(vv, S, Wob, bo, out);
}
